// SAM3E_20151986553018
// MI455X (gfx1250) — matrix operand order repaired and checked by reading, not compiled, not run on hardware
//
#include <hip/hip_runtime.h>

// Shapes from the reference: B=4096, N=128, D=64
#define NB 128   // rows per batch (N)
#define DD 64    // feature dim (D)

typedef _Float16 v16h __attribute__((ext_vector_type(16)));
typedef _Float16 h8   __attribute__((ext_vector_type(8)));
typedef float    v8f  __attribute__((ext_vector_type(8)));
typedef float    f4   __attribute__((ext_vector_type(4)));

// LDS row strides (in halves), padded to de-conflict the 64x4B banks while
// keeping every b128 fragment load 16-byte aligned.
#define LDF   72   // sF  : 128 x 64  f16 (A operand, row-major)
#define LDFT 136   // sFT : 64  x 128 f16 (F transposed, B operand of S@F)
#define LDW   72   // sKw/sQw : 64 x 64 f16
#define LDKF  72   // sKF : 128 x 64 f16
#define LDSS 136   // sS  : 128 x 128 f16

// ---- WMMA fragment loaders (ISA 7.12.2 layouts, wave32) -------------------
// A fragment, 16x32 f16, from row-major LDS matrix `m` (stride ld halves),
// tile top-left (r0, k0). Lane L holds row r0+(L%16); element e:
//   e in [0,8):  K = k0 + 8*(L/16) + e        (two contiguous 16B loads)
//   e in [8,16): K = k0 + 16 + 8*(L/16) + (e-8)
__device__ inline v16h load_a_frag(const _Float16* m, int ld, int r0, int k0, int lane) {
  int row = r0 + (lane & 15);
  int kb  = k0 + ((lane >> 4) << 3);
  h8 lo = *(const h8*)(m + row * ld + kb);
  h8 hi = *(const h8*)(m + row * ld + kb + 16);
  v16h a;
#pragma unroll
  for (int i = 0; i < 8; ++i) { a[i] = lo[i]; a[8 + i] = hi[i]; }
  return a;
}

// B fragment, 32x16 f16, where B[k][c] = R[c0+c][k] for a row-major LDS
// matrix R (stride ld). Lane L holds column c = L%16; elements hold
// K = k0 + 8*(L/16) + e (e<8), k0 + 16 + 8*(L/16) + (e-8) (e>=8) -> two b128 loads from row c0+c.
__device__ inline v16h load_bt_frag(const _Float16* m, int ld, int c0, int k0, int lane) {
  int row = c0 + (lane & 15);
  int kb  = k0 + ((lane >> 4) << 3);
  h8 lo = *(const h8*)(m + row * ld + kb);
  h8 hi = *(const h8*)(m + row * ld + kb + 16);
  v16h b;
#pragma unroll
  for (int i = 0; i < 8; ++i) { b[i] = lo[i]; b[8 + i] = hi[i]; }
  return b;
}

__device__ inline v8f wmma_f16(v16h a, v16h b, v8f c) {
  return __builtin_amdgcn_wmma_f32_16x16x32_f16(
      /*neg_a=*/false, a, /*neg_b=*/false, b,
      /*c_mod=*/(short)0, c, /*reuse_a=*/false, /*reuse_b=*/false);
}

// One workgroup (8 wave32) per batch. Wave w owns row stripe [16w, 16w+16).
__global__ __launch_bounds__(256)
void SAM3E_fastattn_wmma(const float* __restrict__ F,
                         const float* __restrict__ Kw,
                         const float* __restrict__ Qw,
                         float* __restrict__ out)
{
  __shared__ __align__(16) _Float16 sF [NB * LDF];   // F_b  as f16
  __shared__ __align__(16) _Float16 sFT[DD * LDFT];  // F_b^T as f16
  __shared__ __align__(16) _Float16 sKw[DD * LDW];   // K_w  as f16
  __shared__ __align__(16) _Float16 sQw[DD * LDW];   // Q_w  as f16
  __shared__ __align__(16) _Float16 sKF[NB * LDKF];  // KF = F @ Kw^T (f16)
  __shared__ __align__(16) _Float16 sS [NB * LDSS];  // S  = F @ KF^T (f16)

  const int b    = blockIdx.x;
  const int t    = threadIdx.x;
  const int lane = t & 31;
  const int wave = t >> 5;
  const float* Fb = F + (size_t)b * (NB * DD);

  // ---- Phase 0: stage F_b (row-major + transposed) and weights into LDS ---
  // 128x64 f32 = 2048 float4 chunks, 8 per thread (global_load_b128 each).
#pragma unroll
  for (int i = 0; i < 8; ++i) {
    int c   = t + i * 256;
    int row = c >> 4;            // 16 float4 per row
    int c4  = (c & 15) << 2;
    f4 v = *(const f4*)(Fb + row * DD + c4);
    _Float16 h0 = (_Float16)v.x, h1 = (_Float16)v.y,
             h2 = (_Float16)v.z, h3 = (_Float16)v.w;
    _Float16* p = &sF[row * LDF + c4];
    p[0] = h0; p[1] = h1; p[2] = h2; p[3] = h3;      // 8B ds_store
    sFT[(c4 + 0) * LDFT + row] = h0;
    sFT[(c4 + 1) * LDFT + row] = h1;
    sFT[(c4 + 2) * LDFT + row] = h2;
    sFT[(c4 + 3) * LDFT + row] = h3;
  }
  // Weights: 64x64 f32 = 1024 float4 each, 4 per thread.
#pragma unroll
  for (int i = 0; i < 4; ++i) {
    int c   = t + i * 256;
    int row = c >> 4;
    int c4  = (c & 15) << 2;
    f4 k = *(const f4*)(Kw + row * DD + c4);
    f4 q = *(const f4*)(Qw + row * DD + c4);
    _Float16* pk = &sKw[row * LDW + c4];
    pk[0] = (_Float16)k.x; pk[1] = (_Float16)k.y;
    pk[2] = (_Float16)k.z; pk[3] = (_Float16)k.w;
    _Float16* pq = &sQw[row * LDW + c4];
    pq[0] = (_Float16)q.x; pq[1] = (_Float16)q.y;
    pq[2] = (_Float16)q.z; pq[3] = (_Float16)q.w;
  }
  __syncthreads();

  const int r0 = wave * 16;

  // ---- Phase 1: KF = F @ Kw^T (to LDS, f16) and Q = F @ Qw^T (registers) --
  v8f qacc[4];
#pragma unroll
  for (int ct = 0; ct < 4; ++ct) {
    v8f kf = {};
    v8f qa = {};
#pragma unroll
    for (int kk = 0; kk < 2; ++kk) {
      v16h a  = load_a_frag(sF, LDF, r0, kk * 32, lane);
      v16h bk = load_bt_frag(sKw, LDW, ct * 16, kk * 32, lane);  // B[d][e]=Kw[e][d]
      v16h bq = load_bt_frag(sQw, LDW, ct * 16, kk * 32, lane);
      kf = wmma_f16(a, bk, kf);
      qa = wmma_f16(a, bq, qa);
    }
    qacc[ct] = qa;
    // C/D layout: lane holds col = L%16; reg r -> row r + 8*(L/16).
#pragma unroll
    for (int r = 0; r < 8; ++r) {
      int mrow = r0 + r + ((lane >> 4) << 3);
      sKF[mrow * LDKF + ct * 16 + (lane & 15)] = (_Float16)kf[r];
    }
  }
  __syncthreads();

  // ---- Phase 2: S = F @ KF^T (128x128, to LDS f16) ------------------------
#pragma unroll
  for (int ct = 0; ct < 8; ++ct) {
    v8f acc = {};
#pragma unroll
    for (int kk = 0; kk < 2; ++kk) {
      v16h a  = load_a_frag(sF, LDF, r0, kk * 32, lane);
      v16h bb = load_bt_frag(sKF, LDKF, ct * 16, kk * 32, lane); // B[d][m]=KF[m][d]
      acc = wmma_f16(a, bb, acc);
    }
#pragma unroll
    for (int r = 0; r < 8; ++r) {
      int mrow = r0 + r + ((lane >> 4) << 3);
      sS[mrow * LDSS + ct * 16 + (lane & 15)] = (_Float16)acc[r];
    }
  }
  __syncthreads();

  // ---- Phase 3: P = S @ F (K=128), fused epilogue out = F.*P + Q ----------
  float* outb = out + (size_t)b * (NB * DD);
#pragma unroll
  for (int ct = 0; ct < 4; ++ct) {
    v8f acc = {};
#pragma unroll
    for (int kk = 0; kk < 4; ++kk) {
      v16h a  = load_a_frag(sS, LDSS, r0, kk * 32, lane);
      v16h bb = load_bt_frag(sFT, LDFT, ct * 16, kk * 32, lane); // B[m][e]=F[m][e]=FT[e][m]
      acc = wmma_f16(a, bb, acc);
    }
#pragma unroll
    for (int r = 0; r < 8; ++r) {
      int mrow = r0 + r + ((lane >> 4) << 3);
      int col  = ct * 16 + (lane & 15);
      float f  = Fb[mrow * DD + col];            // exact f32 F (L2-hot reload)
      outb[mrow * DD + col] = f * acc[r] + qacc[ct][r];
    }
  }
}

extern "C" void kernel_launch(void* const* d_in, const int* in_sizes, int n_in,
                              void* d_out, int out_size, void* d_ws, size_t ws_size,
                              hipStream_t stream) {
  const float* F  = (const float*)d_in[0];   // (B,128,64) f32
  const float* Kw = (const float*)d_in[1];   // (64,64) f32
  const float* Qw = (const float*)d_in[2];   // (64,64) f32
  float* out = (float*)d_out;                // (B,128,64) f32
  int B = in_sizes[0] / (NB * DD);
  hipLaunchKernelGGL(SAM3E_fastattn_wmma, dim3(B), dim3(256), 0, stream,
                     F, Kw, Qw, out);
}
